// FCOS_RT_62723702391132
// MI455X (gfx1250) — matrix operand order repaired and checked by reading, not compiled, not run on hardware
//
#include <hip/hip_runtime.h>
#include <hip/hip_bf16.h>
#include <math.h>

// ---------------------------------------------------------------------------
// FCOS head for MI455X (gfx1250): batch-0-only implicit-GEMM convolutions on
// v_wmma_f32_16x16x32_bf16.  bf16 weights/activations (converted once),
// 64Mx64N block tile, 64-deep K-steps (8 WMMA/wave between barriers), b128
// LDS fragment loads, TDM (tensor_load_to_lds) weight staging for 1x1 convs,
// fused bias/residual/leaky epilogues, decode, greedy NMS.
// ---------------------------------------------------------------------------

typedef __attribute__((ext_vector_type(16))) __bf16 v16bf;
typedef __attribute__((ext_vector_type(8)))  __bf16 bf16x8;
typedef __attribute__((ext_vector_type(8)))  float  v8f;

#if defined(__HIP_DEVICE_COMPILE__) && __has_builtin(__builtin_amdgcn_tensor_load_to_lds) && __has_builtin(__builtin_amdgcn_s_wait_tensorcnt)
#define HAVE_TDM 1
typedef unsigned int uint32x4 __attribute__((ext_vector_type(4)));
typedef int          int32x8  __attribute__((ext_vector_type(8)));
typedef int          int32x4  __attribute__((ext_vector_type(4)));
#endif

#define NUM_CLASSES 80
#define NANCH 8400
#define IMGF 640.0f

__device__ __forceinline__ __bf16 f2bf(float f) {
  unsigned int u = __float_as_uint(f);
  unsigned int r = (u + 0x7FFFu + ((u >> 16) & 1u)) >> 16;   // RNE
  unsigned short hs = (unsigned short)r;
  __bf16 b;
  __builtin_memcpy(&b, &hs, sizeof(b));
  return b;
}
__device__ __forceinline__ float bf2f(__bf16 b) {
  unsigned short hs;
  __builtin_memcpy(&hs, &b, sizeof(hs));
  return __uint_as_float((unsigned int)hs << 16);
}
__device__ __forceinline__ __bf16 bfz() {
  unsigned short z = 0; __bf16 b;
  __builtin_memcpy(&b, &z, sizeof(b));
  return b;
}
__device__ __forceinline__ float sigm(float x) { return 1.0f / (1.0f + expf(-x)); }

__device__ __forceinline__ v16bf cat8(bf16x8 lo, bf16x8 hi) {
  return __builtin_shufflevector(lo, hi, 0, 1, 2, 3, 4, 5, 6, 7,
                                          8, 9, 10, 11, 12, 13, 14, 15);
}

// ---------------------------------------------------------------------------
__global__ void cvt_f32_bf16_kernel(const float* __restrict__ src,
                                    __bf16* __restrict__ dst, int n)
{
  int i = blockIdx.x * blockDim.x + threadIdx.x;
  if (i < n) dst[i] = f2bf(src[i]);
}

// ---------------------------------------------------------------------------
// Implicit-GEMM conv:  out[co][m] = act( sum_k A[m][k]*B[k][co] + bias + addend )
//   M = H*W, K = Cin*KS*KS (plane-major: k = plane*Cin + ci), N = Cout.
//   Block: 128 threads = 4 waves; block tile 64M x 64N; wave w owns rows
//   [blockIdx.x*64 + w*16, +16) across 4 N sub-tiles: per 64-deep K-step it
//   builds 2 A fragments + 8 B fragments and issues 8 WMMAs.
//   Requires Cin % 64 == 0 (128/256/512 here), so K-steps never straddle a
//   3x3 plane.  LDS rows padded to 72 bf16 (144 B): 16-B aligned b128 loads,
//   36-bank row stride -> lane rows hit disjoint 4-bank windows.
// ---------------------------------------------------------------------------
template <int KS>
__global__ __launch_bounds__(128) void conv_wmma_kernel(
    const __bf16* __restrict__ in, const __bf16* __restrict__ wgt,
    const float* __restrict__ bias, const __bf16* __restrict__ addend,
    __bf16* __restrict__ out,
    int Cin, int Cout, int H, int W, int act)
{
  __shared__ alignas(16) __bf16 As[64][72];   // 64 M x 64 K (+8 pad)
  __shared__ alignas(16) __bf16 Bt[64][72];   // 64 N x 64 K (+8 pad), transposed

  const int M    = H * W;
  const int tid  = threadIdx.x;
  const int lane = tid & 31;
  const int wave = tid >> 5;
  const int hf   = lane >> 4;
  const int l15  = lane & 15;
  const int mB   = blockIdx.x * 64;
  const int n0   = blockIdx.y * 64;
  const int Ktot = Cin * KS * KS;

  // per-element staging coords (3x3 path)
  const int mtA = tid & 63;
  const int kA0 = tid >> 6;
  const int mA  = mB + mtA;
  const bool mOk = (mA < M);
  int yA = 0, xA = 0;
  if (KS == 3 && mOk) { yA = mA / W; xA = mA - yA * W; }

  // vectorized staging coords (1x1 path): b128 runs of 8 bf16
  const int chunk8 = (tid & 7) * 8;
  const int lane16 = tid >> 3;              // 0..15

  v8f acc[4] = {v8f{}, v8f{}, v8f{}, v8f{}};

  for (int k0 = 0; k0 < Ktot; k0 += 64) {
    const int plane  = k0 / Cin;            // exact: Cin % 64 == 0
    const int ciBase = k0 - plane * Cin;

    // ---------------- stage A tile (64M x 64K) ----------------
    if (KS == 1) {
      const bool ok = (mB + chunk8) < M;    // M % 8 == 0 -> whole run valid
#pragma unroll
      for (int i = 0; i < 4; ++i) {
        const int kk = lane16 + 16 * i;
        bf16x8 v = {};
        if (ok) v = *(const bf16x8*)(in + (size_t)(ciBase + kk) * M + mB + chunk8);
#pragma unroll
        for (int e = 0; e < 8; ++e) As[chunk8 + e][kk] = v[e];
      }
    } else {
      const int iy = yA + plane / KS - (KS >> 1);
      const int ix = xA + plane % KS - (KS >> 1);
      const bool ok = mOk && (unsigned)iy < (unsigned)H && (unsigned)ix < (unsigned)W;
      const __bf16* pin = in + (size_t)(ciBase + kA0) * M + iy * W + ix;
      if (ok && ciBase + 64 < Cin)
        __builtin_prefetch(pin + (size_t)64 * M, 0, 1);    // next K-chunk
#pragma unroll
      for (int i = 0; i < 32; ++i)
        As[mtA][kA0 + 2 * i] = ok ? pin[(size_t)(2 * i) * M] : bfz();
    }

    // ---------------- stage B tile transposed (64N x 64K) ----------------
    if (KS == 1) {
#if defined(HAVE_TDM)
      if (wave == 0) {
        // TDM 2D copy: 64 rows x 64 bf16, global stride Cin elements,
        // LDS pad 4 DWORDs every 32 DWORDs -> 72-element padded rows,
        // tensor_dim1 = Cout - n0 zero-fills the N tail (OOB reads -> 0).
        const unsigned long long ga =
            (unsigned long long)(size_t)(wgt + (size_t)n0 * Cin + ciBase);
        const unsigned ldsA = (unsigned)(size_t)&Bt[0][0];
        int rows = Cout - n0; if (rows > 64) rows = 64;
        uint32x4 g0 = { 1u,                                   // count=1
                        ldsA,                                 // lds_addr
                        (unsigned)(ga & 0xffffffffu),
                        (unsigned)((ga >> 32) & 0x1ffffffu) | (2u << 30) }; // type=2
        int32x8 g1 = { (1 << 16) | (1 << 20) | (4 << 22) | (3 << 25),
                       //  data_size=2B | pad_en | interval=32dw | amount=4dw
                       64 << 16,                 // tensor_dim0 = 64     [63:48]
                       (rows & 0xffff) << 16,    // tensor_dim1          [95:80]
                       64 << 16,                 // tile_dim0 = 64       [127:112]
                       64,                       // tile_dim1 = 64       [143:128]
                       Cin,                      // tensor_dim0_stride   [191:160]
                       0, 0 };
        int32x4 gz4 = {};
        int32x8 gz8 = {};
        __builtin_amdgcn_tensor_load_to_lds(g0, g1, gz4, gz4, gz8, 0);
        __builtin_amdgcn_s_wait_tensorcnt((short)0);
      }
#else
#pragma unroll
      for (int i = 0; i < 4; ++i) {
        const int nn = lane16 + 16 * i;
        const int co = n0 + nn;
        bf16x8 v = {};
        if (co < Cout) v = *(const bf16x8*)(wgt + (size_t)co * Cin + ciBase + chunk8);
        *(bf16x8*)&Bt[nn][chunk8] = v;
      }
#endif
    } else {
      const int kB  = tid & 63;
      const int nB0 = tid >> 6;
      const int ci  = ciBase + kB;
#pragma unroll
      for (int i = 0; i < 32; ++i) {
        const int nn = nB0 + 2 * i;
        const int co = n0 + nn;
        Bt[nn][kB] = (co < Cout)
            ? wgt[((size_t)co * Cin + ci) * (KS * KS) + plane] : bfz();
      }
    }
    __syncthreads();

    // -------- 2 sub-chunks x (1 A frag + 4 B frags + 4 WMMAs) --------
#pragma unroll
    for (int s = 0; s < 2; ++s) {
      const bf16x8 alo = *(const bf16x8*)&As[wave * 16 + l15][s * 32 + hf * 8];
      const bf16x8 ahi = *(const bf16x8*)&As[wave * 16 + l15][s * 32 + 16 + hf * 8];
      const v16bf a = cat8(alo, ahi);
#pragma unroll
      for (int nt = 0; nt < 4; ++nt) {
        const bf16x8 blo = *(const bf16x8*)&Bt[nt * 16 + l15][s * 32 + hf * 8];
        const bf16x8 bhi = *(const bf16x8*)&Bt[nt * 16 + l15][s * 32 + 16 + hf * 8];
        const v16bf b = cat8(blo, bhi);
        acc[nt] = __builtin_amdgcn_wmma_f32_16x16x32_bf16(
            false, a, false, b, (short)0, acc[nt], false, false);
      }
    }
    __syncthreads();
  }

  // ---- epilogue: bias + optional residual + optional leaky ReLU(0.1) ----
#pragma unroll
  for (int nt = 0; nt < 4; ++nt) {
    const int co = n0 + nt * 16 + l15;
    if (co >= Cout) continue;
    const float bco = bias[co];
#pragma unroll
    for (int r = 0; r < 8; ++r) {
      const int m = mB + wave * 16 + hf * 8 + r;   // C/D layout: VGPR r -> M
      if (m < M) {
        float v = acc[nt][r] + bco;
        if (addend) v += bf2f(addend[(size_t)co * M + m]);
        if (act)    v = v > 0.0f ? v : 0.1f * v;
        out[(size_t)co * M + m] = f2bf(v);
      }
    }
  }
}

// ---------------------------------------------------------------------------
__global__ void upsample2x_kernel(const __bf16* __restrict__ src,
                                  __bf16* __restrict__ dst, int C, int H, int W)
{
  const int WO = W * 2, HO = H * 2;
  const int tot = C * HO * WO;
  int idx = blockIdx.x * blockDim.x + threadIdx.x;
  if (idx >= tot) return;
  const int x = idx % WO;
  const int y = (idx / WO) % HO;
  const int c = idx / (WO * HO);
  dst[idx] = src[(size_t)c * H * W + (y >> 1) * W + (x >> 1)];
}

// ---------------------------------------------------------------------------
// Per-level decode: scores = sqrt(sig(cls)*sig(ctn)); argmax over classes;
// exp box decode + /IMG clip.  Writes the concatenated output tensor.
// ---------------------------------------------------------------------------
__global__ void decode_kernel(const __bf16* __restrict__ cls,
                              const __bf16* __restrict__ reg,
                              const __bf16* __restrict__ ctn,
                              float* __restrict__ out,
                              int M, int W, float stride, int aOff)
{
  int m = blockIdx.x * blockDim.x + threadIdx.x;
  if (m >= M) return;
  const float cs = sigm(bf2f(ctn[m]));
  float bestv = -1.0f; int bestc = 0;
  for (int c = 0; c < NUM_CLASSES; ++c) {
    float s = sqrtf(sigm(bf2f(cls[(size_t)c * M + m])) * cs);
    if (s > bestv) { bestv = s; bestc = c; }
  }
  const float gx = (float)(m % W), gy = (float)(m / W);
  const float inv = stride * (1.0f / IMGF);
  float x1 = (gx - expf(bf2f(reg[0 * M + m]))) * inv;
  float y1 = (gy - expf(bf2f(reg[1 * M + m]))) * inv;
  float x2 = (gx + expf(bf2f(reg[2 * M + m]))) * inv;
  float y2 = (gy + expf(bf2f(reg[3 * M + m]))) * inv;
  x1 = fminf(fmaxf(x1, 0.0f), 1.0f);
  y1 = fminf(fmaxf(y1, 0.0f), 1.0f);
  x2 = fminf(fmaxf(x2, 0.0f), 1.0f);
  y2 = fminf(fmaxf(y2, 0.0f), 1.0f);
  const int a = aOff + m;
  out[(size_t)a * 4 + 0] = x1;
  out[(size_t)a * 4 + 1] = y1;
  out[(size_t)a * 4 + 2] = x2;
  out[(size_t)a * 4 + 3] = y2;
  out[4 * NANCH + a] = bestv;
  ((int*)out)[5 * NANCH + a] = bestc;
}

// ---------------------------------------------------------------------------
// Greedy per-class NMS, identical semantics to the reference.
// ---------------------------------------------------------------------------
__global__ __launch_bounds__(256) void nms_kernel(
    float* __restrict__ out, int* __restrict__ order,
    int* __restrict__ keep_s, float* __restrict__ areas, int N)
{
  const float* boxes  = out;
  const float* scores = out + 4 * N;
  const int*   cls    = (const int*)(out + 5 * N);
  const int tid = threadIdx.x;

  for (int i = tid; i < N; i += blockDim.x) {
    const float x1 = boxes[i * 4 + 0], y1 = boxes[i * 4 + 1];
    const float x2 = boxes[i * 4 + 2], y2 = boxes[i * 4 + 3];
    areas[i] = (x2 - x1) * (y2 - y1);
    const float si = scores[i];
    int rank = 0;
    for (int j = 0; j < N; ++j) {
      const float sj = scores[j];
      rank += (sj > si) || (sj == si && j < i);
    }
    order[rank] = i;
  }
  __syncthreads();
  for (int t = tid; t < N; t += blockDim.x)
    keep_s[t] = scores[order[t]] >= 0.05f ? 1 : 0;
  __syncthreads();

  for (int t = 0; t < N - 1; ++t) {
    __syncthreads();
    if (!keep_s[t]) continue;            // uniform across block
    const int i = order[t];
    const float x1i = boxes[i * 4 + 0], y1i = boxes[i * 4 + 1];
    const float x2i = boxes[i * 4 + 2], y2i = boxes[i * 4 + 3];
    const float ai  = areas[i];
    const int   ci  = cls[i];
    for (int tt = t + 1 + tid; tt < N; tt += blockDim.x) {
      const int j = order[tt];
      if (cls[j] != ci) continue;
      const float xx1 = fmaxf(x1i, boxes[j * 4 + 0]);
      const float yy1 = fmaxf(y1i, boxes[j * 4 + 1]);
      const float xx2 = fminf(x2i, boxes[j * 4 + 2]);
      const float yy2 = fminf(y2i, boxes[j * 4 + 3]);
      const float w = fmaxf(1e-28f, xx2 - xx1);
      const float h = fmaxf(1e-28f, yy2 - yy1);
      const float inter = w * h;
      const float ovr = inter / (ai + areas[j] - inter);
      if (ovr > 0.5f) keep_s[tt] = 0;
    }
  }
  __syncthreads();
  for (int t = tid; t < N; t += blockDim.x)
    ((int*)out)[6 * N + order[t]] = keep_s[t];
}

// ---------------------------------------------------------------------------
static inline void run_cvt(const float* src, __bf16* dst, int n, hipStream_t s) {
  cvt_f32_bf16_kernel<<<(n + 255) / 256, 256, 0, s>>>(src, dst, n);
}

static inline void run_conv(const __bf16* in, const __bf16* w, const float* b,
                            const __bf16* addend, __bf16* out,
                            int Cin, int Cout, int H, int W, int ks, int act,
                            hipStream_t s)
{
  const int M = H * W;
  dim3 grid((M + 63) / 64, (Cout + 63) / 64, 1);
  if (ks == 1)
    conv_wmma_kernel<1><<<grid, 128, 0, s>>>(in, w, b, addend, out, Cin, Cout, H, W, act);
  else
    conv_wmma_kernel<3><<<grid, 128, 0, s>>>(in, w, b, addend, out, Cin, Cout, H, W, act);
}

extern "C" void kernel_launch(void* const* d_in, const int* in_sizes, int n_in,
                              void* d_out, int out_size, void* d_ws, size_t ws_size,
                              hipStream_t stream)
{
  const float* c3        = (const float*)d_in[0];   // (4,128,80,80) -> batch 0
  const float* c4        = (const float*)d_in[1];   // (4,256,40,40)
  const float* c5        = (const float*)d_in[2];   // (4,512,20,20)
  const float* lat1_w    = (const float*)d_in[3];
  const float* lat1_b    = (const float*)d_in[4];
  const float* lat2_w    = (const float*)d_in[5];
  const float* lat2_b    = (const float*)d_in[6];
  const float* lat3_w    = (const float*)d_in[7];
  const float* lat3_b    = (const float*)d_in[8];
  const float* sm1_w     = (const float*)d_in[9];
  const float* sm1_b     = (const float*)d_in[10];
  const float* sm2_w     = (const float*)d_in[11];
  const float* sm2_b     = (const float*)d_in[12];
  const float* sm3_w     = (const float*)d_in[13];
  const float* sm3_b     = (const float*)d_in[14];
  const float* clsh_w    = (const float*)d_in[15];  // (4,256,256,3,3)
  const float* clsh_b    = (const float*)d_in[16];  // (4,256)
  const float* regh_w    = (const float*)d_in[17];
  const float* regh_b    = (const float*)d_in[18];
  const float* cls_det_w = (const float*)d_in[19];
  const float* cls_det_b = (const float*)d_in[20];
  const float* reg_det_w = (const float*)d_in[21];
  const float* reg_det_b = (const float*)d_in[22];
  const float* ctn_det_w = (const float*)d_in[23];
  const float* ctn_det_b = (const float*)d_in[24];
  (void)in_sizes; (void)n_in; (void)out_size; (void)ws_size;

  // ---- workspace carve-out: bf16 region (all sizes multiples of 128) ----
  __bf16* bw = (__bf16*)d_ws;
  __bf16* c3b = bw;  bw += 819200;     // 128*6400
  __bf16* c4b = bw;  bw += 409600;     // 256*1600
  __bf16* c5b = bw;  bw += 204800;     // 512*400
  __bf16* wl1 = bw;  bw += 32768;      // lat1 256x128
  __bf16* wl2 = bw;  bw += 65536;      // lat2 256x256
  __bf16* wl3 = bw;  bw += 131072;     // lat3 256x512
  __bf16* ws1 = bw;  bw += 589824;     // sm1 256x256x9
  __bf16* ws2 = bw;  bw += 589824;
  __bf16* ws3 = bw;  bw += 589824;
  __bf16* wch = bw;  bw += 2359296;    // cls head 4x(256x256x9)
  __bf16* wrh = bw;  bw += 2359296;    // reg head
  __bf16* wcd = bw;  bw += 20480;      // cls det 80x256
  __bf16* wrd = bw;  bw += 1024;       // reg det 4x256
  __bf16* wtd = bw;  bw += 256;        // ctn det 1x256
  __bf16* t5  = bw;  bw += 102400;
  __bf16* u5  = bw;  bw += 409600;
  __bf16* p5  = bw;  bw += 102400;
  __bf16* t4  = bw;  bw += 409600;
  __bf16* u4  = bw;  bw += 1638400;
  __bf16* p4  = bw;  bw += 409600;
  __bf16* t3  = bw;  bw += 1638400;
  __bf16* p3  = bw;  bw += 1638400;
  __bf16* h0  = bw;  bw += 1638400;
  __bf16* h1  = bw;  bw += 1638400;
  __bf16* clsb = bw; bw += 512000;     // 80*6400
  __bf16* regb = bw; bw += 25600;      // 4*6400
  __bf16* ctnb = bw; bw += 6400;
  int*   order = (int*)bw;
  int*   keeps = order + NANCH;
  float* areas = (float*)(keeps + NANCH);

  // ---- one-time f32 -> bf16 conversion of weights + batch-0 activations ----
  run_cvt(c3, c3b, 819200, stream);
  run_cvt(c4, c4b, 409600, stream);
  run_cvt(c5, c5b, 204800, stream);
  run_cvt(lat1_w, wl1, 32768, stream);
  run_cvt(lat2_w, wl2, 65536, stream);
  run_cvt(lat3_w, wl3, 131072, stream);
  run_cvt(sm1_w, ws1, 589824, stream);
  run_cvt(sm2_w, ws2, 589824, stream);
  run_cvt(sm3_w, ws3, 589824, stream);
  run_cvt(clsh_w, wch, 2359296, stream);
  run_cvt(regh_w, wrh, 2359296, stream);
  run_cvt(cls_det_w, wcd, 20480, stream);
  run_cvt(reg_det_w, wrd, 1024, stream);
  run_cvt(ctn_det_w, wtd, 256, stream);

  // ---- FPN (batch 0 only; outputs beyond [0] are never used downstream) ----
  run_conv(c5b, wl3, lat3_b, nullptr, t5, 512, 256, 20, 20, 1, 0, stream);
  {
    const int tot = 256 * 40 * 40;
    upsample2x_kernel<<<(tot + 255) / 256, 256, 0, stream>>>(t5, u5, 256, 20, 20);
  }
  run_conv(t5, ws3, sm3_b, nullptr, p5, 256, 256, 20, 20, 3, 0, stream);
  run_conv(c4b, wl2, lat2_b, u5, t4, 256, 256, 40, 40, 1, 0, stream);
  {
    const int tot = 256 * 80 * 80;
    upsample2x_kernel<<<(tot + 255) / 256, 256, 0, stream>>>(t4, u4, 256, 40, 40);
  }
  run_conv(t4, ws2, sm2_b, nullptr, p4, 256, 256, 40, 40, 3, 0, stream);
  run_conv(c3b, wl1, lat1_b, u4, t3, 128, 256, 80, 80, 1, 0, stream);
  run_conv(t3, ws1, sm1_b, nullptr, p3, 256, 256, 80, 80, 3, 0, stream);

  // ---- heads + decode per level ----
  const __bf16* plv[3] = { p3, p4, p5 };
  const int    hw[3]   = { 80, 40, 20 };
  const float  strd[3] = { 8.0f, 16.0f, 32.0f };
  const int    aoff[3] = { 0, 6400, 8000 };
  float* outf = (float*)d_out;

  for (int L = 0; L < 3; ++L) {
    const int H = hw[L], M = H * H;

    // classification tower (4x conv3x3 + leaky) -> 1x1 det
    const __bf16* cur = plv[L];
    for (int i = 0; i < 4; ++i) {
      __bf16* dst = (i & 1) ? h1 : h0;
      run_conv(cur, wch + (size_t)i * 256 * 256 * 9, clsh_b + i * 256,
               nullptr, dst, 256, 256, H, H, 3, 1, stream);
      cur = dst;
    }
    run_conv(cur, wcd, cls_det_b, nullptr, clsb, 256, NUM_CLASSES, H, H, 1, 0, stream);

    // regression tower -> box (4ch) + centerness (1ch) 1x1 dets
    cur = plv[L];
    for (int i = 0; i < 4; ++i) {
      __bf16* dst = (i & 1) ? h1 : h0;
      run_conv(cur, wrh + (size_t)i * 256 * 256 * 9, regh_b + i * 256,
               nullptr, dst, 256, 256, H, H, 3, 1, stream);
      cur = dst;
    }
    run_conv(cur, wrd, reg_det_b, nullptr, regb, 256, 4, H, H, 1, 0, stream);
    run_conv(cur, wtd, ctn_det_b, nullptr, ctnb, 256, 1, H, H, 1, 0, stream);

    decode_kernel<<<(M + 255) / 256, 256, 0, stream>>>(
        clsb, regb, ctnb, outf, M, H, strd[L], aoff[L]);
  }

  // ---- greedy NMS over all 8400 anchors ----
  nms_kernel<<<1, 256, 0, stream>>>(outf, order, keeps, areas, NANCH);
}
